// PointNet2Backbone_8091718385923
// MI455X (gfx1250) — matrix operand order repaired and checked by reading, not compiled, not run on hardware
//
#include <hip/hip_runtime.h>
#include <hip/hip_bf16.h>

typedef __attribute__((ext_vector_type(16))) _Float16 v16h;
typedef __attribute__((ext_vector_type(8)))  float    v8f;
typedef _Float16 h16;

#define BB    2
#define NFULL 16384
#define NSUB  2048

static inline int pad32(int x) { return (x + 31) & ~31; }

// ---------------------------------------------------------------------------
// Farthest point sampling: one workgroup per batch. dist[] lives in global
// (L2-resident), argmax via LDS tree reduction, tie-break = lowest index to
// match jnp.argmax.
// ---------------------------------------------------------------------------
__global__ void fps_kernel(const float* __restrict__ xyz, int n, int npoint,
                           int* __restrict__ out, float* __restrict__ dist) {
  __shared__ float sval[256];
  __shared__ int   sidx[256];
  __shared__ float lastp[3];
  const int b = blockIdx.x, tid = threadIdx.x;
  const float* xb = xyz + (size_t)b * n * 3;
  float* db = dist + (size_t)b * n;
  for (int i = tid; i < n; i += 256) db[i] = 1e10f;
  if (tid == 0) {
    out[b * npoint] = 0;
    lastp[0] = xb[0]; lastp[1] = xb[1]; lastp[2] = xb[2];
  }
  __syncthreads();
  for (int it = 1; it < npoint; ++it) {
    const float lx = lastp[0], ly = lastp[1], lz = lastp[2];
    float best = -1.0f; int bi = n;
    for (int i = tid; i < n; i += 256) {
      float dx = xb[i * 3 + 0] - lx;
      float dy = xb[i * 3 + 1] - ly;
      float dz = xb[i * 3 + 2] - lz;
      float d  = dx * dx + dy * dy + dz * dz;
      float nd = fminf(db[i], d);
      db[i] = nd;
      if (nd > best) { best = nd; bi = i; }   // strict > keeps lowest index
    }
    sval[tid] = best; sidx[tid] = bi;
    __syncthreads();
    for (int s = 128; s > 0; s >>= 1) {
      if (tid < s) {
        float ov = sval[tid + s]; int oi = sidx[tid + s];
        if (ov > sval[tid] || (ov == sval[tid] && oi < sidx[tid])) {
          sval[tid] = ov; sidx[tid] = oi;
        }
      }
      __syncthreads();
    }
    if (tid == 0) {
      int sel = sidx[0];
      out[b * npoint + it] = sel;
      lastp[0] = xb[sel * 3 + 0]; lastp[1] = xb[sel * 3 + 1]; lastp[2] = xb[sel * 3 + 2];
    }
    __syncthreads();
  }
}

// ---------------------------------------------------------------------------
// Generic gather: dst[b, i, c] = src[b, idx[b,i], c]
// ---------------------------------------------------------------------------
__global__ void gather_kernel(const float* __restrict__ src, const int* __restrict__ idx,
                              float* __restrict__ dst, int n_src, int m, int C) {
  int t = blockIdx.x * blockDim.x + threadIdx.x;
  int total = BB * m * C;
  if (t >= total) return;
  int c = t % C; int r = t / C; int i = r % m; int b = r / m;
  int s = idx[b * m + i];
  dst[t] = src[((size_t)b * n_src + s) * C + c];
}

// ---------------------------------------------------------------------------
// Ball query: scan in ascending index order, keep first nsample inside radius
// (== sort-then-truncate of the reference); pad with first hit, else 0.
// ---------------------------------------------------------------------------
__global__ void ball_query_kernel(const float* __restrict__ xyz,
                                  const float* __restrict__ new_xyz,
                                  int* __restrict__ idx, int n_src, int np,
                                  float r2, int ns) {
  int t = blockIdx.x * blockDim.x + threadIdx.x;
  if (t >= BB * np) return;
  int b = t / np;
  const float* xb = xyz + (size_t)b * n_src * 3;
  float cx = new_xyz[t * 3 + 0], cy = new_xyz[t * 3 + 1], cz = new_xyz[t * 3 + 2];
  int* o = idx + (size_t)t * ns;
  int cnt = 0;
  for (int i = 0; i < n_src && cnt < ns; ++i) {
    float dx = xb[i * 3 + 0] - cx, dy = xb[i * 3 + 1] - cy, dz = xb[i * 3 + 2] - cz;
    if (dx * dx + dy * dy + dz * dz < r2) { o[cnt++] = i; }
  }
  int fill = (cnt > 0) ? o[0] : 0;
  for (int s = cnt; s < ns; ++s) o[s] = fill;
}

// ---------------------------------------------------------------------------
// Build grouped A-matrix rows (f16, zero-padded to Kp): [g_xyz(3), feats(Cin)]
// ---------------------------------------------------------------------------
__global__ void group_gather_kernel(const float* __restrict__ xyz,
                                    const float* __restrict__ feats,
                                    const float* __restrict__ new_xyz,
                                    const int* __restrict__ idx,
                                    h16* __restrict__ A,
                                    int n_src, int np, int ns, int Cin, int Kp) {
  int m = blockIdx.x * blockDim.x + threadIdx.x;
  int total = BB * np * ns;
  if (m >= total) return;
  int cj = m / ns;            // b*np + j
  int b  = cj / np;
  int i  = idx[m];
  h16* row = A + (size_t)m * Kp;
  const float* pc = new_xyz + (size_t)cj * 3;
  const float* pp = xyz + ((size_t)b * n_src + i) * 3;
  row[0] = (h16)(pp[0] - pc[0]);
  row[1] = (h16)(pp[1] - pc[1]);
  row[2] = (h16)(pp[2] - pc[2]);
  const float* pf = feats + ((size_t)b * n_src + i) * Cin;
  for (int c = 0; c < Cin; ++c) row[3 + c] = (h16)pf[c];
  for (int k = 3 + Cin; k < Kp; ++k) row[k] = (h16)0.f;
}

// ---------------------------------------------------------------------------
// three_nn + interpolation + concat into f16 A-matrix rows:
// [interp(f2, 3-NN weights)(C2), f1(C1), zero-pad to Kp]
// ---------------------------------------------------------------------------
__global__ void interp_concat_kernel(const float* __restrict__ xyz1,
                                     const float* __restrict__ xyz2,
                                     const float* __restrict__ f2,
                                     const float* __restrict__ f1,
                                     h16* __restrict__ A,
                                     int n1, int n2, int C2, int C1, int Kp) {
  int t = blockIdx.x * blockDim.x + threadIdx.x;
  if (t >= BB * n1) return;
  int b = t / n1;
  float px = xyz1[t * 3 + 0], py = xyz1[t * 3 + 1], pz = xyz1[t * 3 + 2];
  const float* x2 = xyz2 + (size_t)b * n2 * 3;
  float d0 = 1e30f, d1 = 1e30f, d2v = 1e30f;
  int i0 = 0, i1 = 0, i2 = 0;
  for (int j = 0; j < n2; ++j) {
    float dx = x2[j * 3 + 0] - px, dy = x2[j * 3 + 1] - py, dz = x2[j * 3 + 2] - pz;
    float d = dx * dx + dy * dy + dz * dz;
    if (d < d0)       { d2v = d1; i2 = i1; d1 = d0; i1 = i0; d0 = d; i0 = j; }
    else if (d < d1)  { d2v = d1; i2 = i1; d1 = d;  i1 = j; }
    else if (d < d2v) { d2v = d;  i2 = j; }
  }
  float w0 = 1.f / (d0 + 1e-8f), w1 = 1.f / (d1 + 1e-8f), w2 = 1.f / (d2v + 1e-8f);
  float wsum = w0 + w1 + w2;
  w0 /= wsum; w1 /= wsum; w2 /= wsum;
  h16* row = A + (size_t)t * Kp;
  const float* f2b = f2 + (size_t)b * n2 * C2;
  const float* r0 = f2b + (size_t)i0 * C2;
  const float* r1 = f2b + (size_t)i1 * C2;
  const float* r2 = f2b + (size_t)i2 * C2;
  for (int c = 0; c < C2; ++c)
    row[c] = (h16)(w0 * r0[c] + w1 * r1[c] + w2 * r2[c]);
  if (C1 > 0) {
    const float* p1 = f1 + (size_t)t * C1;
    for (int c = 0; c < C1; ++c) row[C2 + c] = (h16)p1[c];
  }
  for (int k = C2 + C1; k < Kp; ++k) row[k] = (h16)0.f;
}

// ---------------------------------------------------------------------------
// Weight packing: W(cout,cin) f32 -> f16 zero-padded rows of length Kp,
// BN folded: effs = scale, effb = b*scale + shift.
// ---------------------------------------------------------------------------
__global__ void pack_weights_kernel(const float* __restrict__ W, const float* __restrict__ b,
                                    const float* __restrict__ s, const float* __restrict__ sh,
                                    h16* __restrict__ Wp, float* __restrict__ effs,
                                    float* __restrict__ effb, int Cout, int Cin, int Kp) {
  int t = blockIdx.x * blockDim.x + threadIdx.x;
  if (t < Cout) { effs[t] = s[t]; effb[t] = b[t] * s[t] + sh[t]; }
  int total = Cout * Kp;
  int stride = gridDim.x * blockDim.x;
  for (int x = t; x < total; x += stride) {
    int k = x % Kp, n = x / Kp;
    Wp[x] = (k < Cin) ? (h16)W[n * Cin + k] : (h16)0.f;
  }
}

// ---------------------------------------------------------------------------
// Register-blocked WMMA GEMM + fused BN/ReLU epilogue.
// A: M x Kp f16 row-major (Kp % 32 == 0). Wp: Ncols x Kp f16 row-major (B^T).
// Each wave owns a 16 x (NT*16) output strip: A fragment loaded once per
// K-chunk and reused across NT v_wmma_f32_16x16x32_f16 issues.
// wave32 A/B use contiguous K order (k = 16*half + s); D layout per CDNA5 ISA 7.12.2. Grids are exact -> EXEC all ones.
// ---------------------------------------------------------------------------
template <int NT, bool OUT16>
__global__ void gemm_bn_relu_kernel(const h16* __restrict__ A, int Kp,
                                    const h16* __restrict__ Wp,
                                    const float* __restrict__ effs,
                                    const float* __restrict__ effb,
                                    void* __restrict__ outp, int ldo) {
  const int mt = blockIdx.x * blockDim.y + threadIdx.y;  // M-tile of this wave
  const int ng = blockIdx.y;                             // group of NT N-tiles
  const int lane = threadIdx.x;
  const int lh = lane & 15;
  const int hs = lane >> 4;                              // lane half: 0 or 1
  const h16* arow = A + ((size_t)(mt * 16 + lh)) * Kp;
  const h16* wrow[NT];
#pragma unroll
  for (int t = 0; t < NT; ++t)
    wrow[t] = Wp + ((size_t)((ng * NT + t) * 16 + lh)) * Kp;
  const v8f vzero = {};
  v8f acc[NT];
#pragma unroll
  for (int t = 0; t < NT; ++t) acc[t] = vzero;
  for (int k0 = 0; k0 < Kp; k0 += 32) {
    v16h av;
#pragma unroll
    for (int j = 0; j < 8; ++j) {
      av[j]     = arow[k0 + hs * 16 + j];       // K = hs*16 + j
      av[j + 8] = arow[k0 + hs * 16 + 8 + j];   // K = hs*16 + 8 + j
    }
#pragma unroll
    for (int t = 0; t < NT; ++t) {
      v16h bv;
#pragma unroll
      for (int j = 0; j < 16; ++j)
        bv[j] = wrow[t][k0 + hs * 16 + j];      // K = hs*16 + j, N = lh
      acc[t] = __builtin_amdgcn_wmma_f32_16x16x32_f16(
          false, av, false, bv, (short)0, acc[t], false, false);
    }
  }
#pragma unroll
  for (int t = 0; t < NT; ++t) {
    const int col = (ng * NT + t) * 16 + lh;
    const float sc = effs[col];
    const float bs = effb[col];
#pragma unroll
    for (int v = 0; v < 8; ++v) {
      int row = mt * 16 + v + hs * 8;           // D layout: VGPR v -> M = v (+8)
      float y = fmaxf(acc[t][v] * sc + bs, 0.f);
      if (OUT16) ((h16*)outp)[(size_t)row * ldo + col] = (h16)y;
      else       ((float*)outp)[(size_t)row * ldo + col] = y;
    }
  }
}

// ---------------------------------------------------------------------------
// Max-pool over nsample consecutive rows -> feature slab [B, np, Ctot] @ coff
// ---------------------------------------------------------------------------
__global__ void maxpool_kernel(const float* __restrict__ G, float* __restrict__ feats,
                               int np, int ns, int Cout, int Ctot, int coff) {
  int t = blockIdx.x * blockDim.x + threadIdx.x;
  int total = BB * np * Cout;
  if (t >= total) return;
  int c = t % Cout; int p = t / Cout;
  const float* g = G + ((size_t)p * ns) * Cout + c;
  float m = g[0];
  for (int s = 1; s < ns; ++s) m = fmaxf(m, g[(size_t)s * Cout]);
  feats[(size_t)p * Ctot + coff + c] = m;
}

// ===========================================================================
// Host orchestration
// ===========================================================================
struct Layer {
  const float *W, *b, *s, *sh;
  int cout, cin, Kp;
  h16* Wp;
  float* effs;
  float* effb;
};

extern "C" void kernel_launch(void* const* d_in, const int* in_sizes, int n_in,
                              void* d_out, int out_size, void* d_ws, size_t ws_size,
                              hipStream_t stream) {
  (void)in_sizes; (void)n_in; (void)out_size; (void)ws_size;
  const float* xyz   = (const float*)d_in[0];
  const float* feats = (const float*)d_in[1];

  // deterministic 256B-aligned arena in d_ws
  char* base = (char*)d_ws;
  size_t off = 0;
  auto alloc = [&](size_t bytes) -> void* {
    void* p = base + off;
    off += (bytes + 255) & ~(size_t)255;
    return p;
  };

  // ---- parse params: (W,b,scale,shift) tuples in pytree order ----
  Layer Ls[19];
  int nl = 0, p = 2;
  auto take = [&](int cout, int cin) -> int {
    Layer L;
    L.W  = (const float*)d_in[p + 0];
    L.b  = (const float*)d_in[p + 1];
    L.s  = (const float*)d_in[p + 2];
    L.sh = (const float*)d_in[p + 3];
    p += 4;
    L.cout = cout; L.cin = cin; L.Kp = pad32(cin);
    L.Wp   = (h16*)alloc((size_t)cout * L.Kp * sizeof(h16));
    L.effs = (float*)alloc((size_t)cout * sizeof(float));
    L.effb = (float*)alloc((size_t)cout * sizeof(float));
    Ls[nl] = L;
    return nl++;
  };
  int sa1[2][2], sa2[2][2], sa3[2][2];
  for (int sc = 0; sc < 2; ++sc) { sa1[sc][0] = take(32, 6);    sa1[sc][1] = take(64, 32);   }
  for (int sc = 0; sc < 2; ++sc) { sa2[sc][0] = take(64, 131);  sa2[sc][1] = take(128, 64);  }
  for (int sc = 0; sc < 2; ++sc) { sa3[sc][0] = take(128, 259); sa3[sc][1] = take(256, 128); }
  int fp3l[2]  = { take(256, 768), take(256, 256) };
  int fp2l[2]  = { take(256, 384), take(256, 256) };
  int fp1l[2]  = { take(256, 259), take(256, 256) };
  int finalL   = take(512, 256);

  for (int i = 0; i < nl; ++i) {
    const Layer& L = Ls[i];
    int total = L.cout * L.Kp;
    pack_weights_kernel<<<(total + 255) / 256, 256, 0, stream>>>(
        L.W, L.b, L.s, L.sh, L.Wp, L.effs, L.effb, L.cout, L.cin, L.Kp);
  }

  // ---- scratch buffers ----
  float* dist    = (float*)alloc((size_t)BB * NFULL * 4);
  int*   fidx0   = (int*)  alloc((size_t)BB * NSUB * 4);
  float* xyz_sub = (float*)alloc((size_t)BB * NSUB * 3 * 4);
  float* f_sub   = (float*)alloc((size_t)BB * NSUB * 3 * 4);
  int*   fidx1   = (int*)  alloc((size_t)BB * 512 * 4);
  float* l1_xyz  = (float*)alloc((size_t)BB * 512 * 3 * 4);
  int*   fidx2   = (int*)  alloc((size_t)BB * 128 * 4);
  float* l2_xyz  = (float*)alloc((size_t)BB * 128 * 3 * 4);
  int*   fidx3   = (int*)  alloc((size_t)BB * 32 * 4);
  float* l3_xyz  = (float*)alloc((size_t)BB * 32 * 3 * 4);
  float* l1_f    = (float*)alloc((size_t)BB * 512 * 128 * 4);
  float* l2_f    = (float*)alloc((size_t)BB * 128 * 256 * 4);
  float* l3_f    = (float*)alloc((size_t)BB * 32 * 512 * 4);
  float* l2_fn   = (float*)alloc((size_t)BB * 128 * 256 * 4);
  float* l1_fn   = (float*)alloc((size_t)BB * 512 * 256 * 4);
  float* l0_f    = (float*)alloc((size_t)BB * NSUB * 256 * 4);
  int*   ballidx = (int*)  alloc((size_t)BB * 512 * 16 * 4);
  h16*   Abuf    = (h16*)  alloc((size_t)BB * NFULL * 256 * sizeof(h16));
  h16*   Hbuf    = (h16*)  alloc((size_t)BB * NSUB * 256 * sizeof(h16));
  float* Gbuf    = (float*)alloc((size_t)BB * 512 * 16 * 64 * 4);

  // M is always a multiple of 64; cout is 32 (NT=2) or a multiple of 64 (NT=4)
  auto gemm = [&](const h16* A, int Kp, const Layer& L, void* out, int ldo,
                  int M, bool out16) {
    dim3 blk(32, 4, 1);
    int ntiles = L.cout / 16;
    if (L.cout % 64 == 0) {
      dim3 grd(M / 64, ntiles / 4, 1);
      if (out16)
        gemm_bn_relu_kernel<4, true><<<grd, blk, 0, stream>>>(A, Kp, L.Wp, L.effs, L.effb, out, ldo);
      else
        gemm_bn_relu_kernel<4, false><<<grd, blk, 0, stream>>>(A, Kp, L.Wp, L.effs, L.effb, out, ldo);
    } else {
      dim3 grd(M / 64, ntiles / 2, 1);
      if (out16)
        gemm_bn_relu_kernel<2, true><<<grd, blk, 0, stream>>>(A, Kp, L.Wp, L.effs, L.effb, out, ldo);
      else
        gemm_bn_relu_kernel<2, false><<<grd, blk, 0, stream>>>(A, Kp, L.Wp, L.effs, L.effb, out, ldo);
    }
  };

  auto run_sa = [&](const float* xyz_in, const float* feats_in, int n_src, int Cin,
                    int np, const float* r2, int* fidx, float* new_xyz,
                    float* feats_out, int Ctot, const int mlps[2][2]) {
    fps_kernel<<<BB, 256, 0, stream>>>(xyz_in, n_src, np, fidx, dist);
    {
      int total = BB * np * 3;
      gather_kernel<<<(total + 255) / 256, 256, 0, stream>>>(xyz_in, fidx, new_xyz, n_src, np, 3);
    }
    int M = BB * np * 16;
    for (int sc = 0; sc < 2; ++sc) {
      const Layer& L0 = Ls[mlps[sc][0]];
      const Layer& L1 = Ls[mlps[sc][1]];
      {
        int total = BB * np;
        ball_query_kernel<<<(total + 127) / 128, 128, 0, stream>>>(
            xyz_in, new_xyz, ballidx, n_src, np, r2[sc], 16);
      }
      {
        int total = BB * np * 16;
        group_gather_kernel<<<(total + 255) / 256, 256, 0, stream>>>(
            xyz_in, feats_in, new_xyz, ballidx, Abuf, n_src, np, 16, Cin, L0.Kp);
      }
      gemm(Abuf, L0.Kp, L0, Hbuf, L0.cout, M, true);
      gemm(Hbuf, L0.cout, L1, Gbuf, L1.cout, M, false);
      {
        int total = BB * np * L1.cout;
        maxpool_kernel<<<(total + 255) / 256, 256, 0, stream>>>(
            Gbuf, feats_out, np, 16, L1.cout, Ctot, sc * L1.cout);
      }
    }
  };

  auto run_fp = [&](const float* xyz1, int n1, const float* xyz2, int n2,
                    const float* f1, int C1, const float* f2, int C2,
                    const int lids[2], float* outf) {
    const Layer& L0 = Ls[lids[0]];
    const Layer& L1 = Ls[lids[1]];
    int total = BB * n1;
    interp_concat_kernel<<<(total + 127) / 128, 128, 0, stream>>>(
        xyz1, xyz2, f2, f1, Abuf, n1, n2, C2, C1, L0.Kp);
    int M = BB * n1;
    gemm(Abuf, L0.Kp, L0, Hbuf, L0.cout, M, true);
    gemm(Hbuf, L0.cout, L1, outf, L1.cout, M, false);
  };

  // ---- stage 0: subsample to 2048 ----
  fps_kernel<<<BB, 256, 0, stream>>>(xyz, NFULL, NSUB, fidx0, dist);
  {
    int total = BB * NSUB * 3;
    gather_kernel<<<(total + 255) / 256, 256, 0, stream>>>(xyz, fidx0, xyz_sub, NFULL, NSUB, 3);
    gather_kernel<<<(total + 255) / 256, 256, 0, stream>>>(feats, fidx0, f_sub, NFULL, NSUB, 3);
  }

  // ---- SA hierarchy ----
  const float r1[2]  = { 0.02f * 0.02f, 0.04f * 0.04f };
  const float r2v[2] = { 0.04f * 0.04f, 0.08f * 0.08f };
  const float r3[2]  = { 0.08f * 0.08f, 0.16f * 0.16f };
  run_sa(xyz_sub, f_sub, NSUB, 3,   512, r1,  fidx1, l1_xyz, l1_f, 128, sa1);
  run_sa(l1_xyz,  l1_f,  512,  128, 128, r2v, fidx2, l2_xyz, l2_f, 256, sa2);
  run_sa(l2_xyz,  l2_f,  128,  256, 32,  r3,  fidx3, l3_xyz, l3_f, 512, sa3);

  // ---- FP hierarchy ----
  run_fp(l2_xyz, 128,  l3_xyz, 32,  l2_f, 256, l3_f,  512, fp3l, l2_fn);
  run_fp(l1_xyz, 512,  l2_xyz, 128, l1_f, 128, l2_fn, 256, fp2l, l1_fn);
  run_fp(xyz_sub, NSUB, l1_xyz, 512, f_sub, 3, l1_fn, 256, fp1l, l0_f);

  // ---- final: interp to full resolution + final conv -> d_out (B,N,512) ----
  {
    int total = BB * NFULL;
    interp_concat_kernel<<<(total + 127) / 128, 128, 0, stream>>>(
        xyz, xyz_sub, l0_f, nullptr, Abuf, NFULL, NSUB, 256, 0, 256);
    gemm(Abuf, 256, Ls[finalL], d_out, 512, BB * NFULL, false);
  }
}
